// GAU_50672024158214
// MI455X (gfx1250) — matrix operand order repaired and checked by reading, not compiled, not run on hardware
//
#include <hip/hip_runtime.h>
#include <stdint.h>

// ---------------------------------------------------------------------------
// GAU (gated attention unit) for MI455X / gfx1250, all GEMMs via
// v_wmma_f32_16x16x32_bf16 (wave32). Shapes: b=4 n=4096 DIM=512 QK=128 H=1024.
// Pipeline:
//   K0  convert x -> bf16; transpose weights -> bf16 [N][K]
//   K1  hid = x@Wh + bh          -> vT (bf16, [b][h][n]) and gate (bf16, [b][n][h])
//   K2  qk  = x@Wqk              -> q,k (bf16, [b][n][128]) with gamma/beta
//   K3  attn = relu(q k^T * s)^2 -> bf16 [b][n][n]   (134 MB, fits 192MB L2)
//   K4  o = (attn @ v) * gate    -> bf16 [b][n][1024]  (async-LDS B staging)
//   K5  out = o@Wo + bo + x      -> f32
// Wave tile: 32(M) x 64(N) = 8 accumulators; block = 8 waves = 256x64 tile.
// ---------------------------------------------------------------------------

typedef __attribute__((ext_vector_type(16))) __bf16 v16bf;
typedef __attribute__((ext_vector_type(8)))  float  v8f;

union FragBF { v16bf v; uint4 q[2]; };
union Pack8  { uint4 q; unsigned short s[8]; };

__device__ __forceinline__ unsigned short f2bf(float f) {
  unsigned int u = __float_as_uint(f);
  unsigned int r = (u + 0x7FFFu + ((u >> 16) & 1u)) >> 16;  // RNE
  return (unsigned short)r;
}
__device__ __forceinline__ float bf2f(unsigned short s) {
  return __uint_as_float(((unsigned int)s) << 16);
}

#define WMMA_BF16(Afrag, Bfrag, Cacc)                                          \
  __builtin_amdgcn_wmma_f32_16x16x32_bf16(false, (Afrag).v, false, (Bfrag).v,  \
                                          (short)0, (Cacc), false, false)

// Each wave computes a 32(M) x 64(N) f32 tile: acc[2][4] of 16x16.
// A  : row-major [M][K] bf16 (as u16), lda in elements.
// BT : [N][K] bf16 (B stored K-contiguous per output column), ldb in elements.
__device__ __forceinline__ void wmma_tile_32x64(
    const unsigned short* __restrict__ A, int lda,
    const unsigned short* __restrict__ BT, int ldb,
    int K, int mRow, int nBase, v8f acc[2][4])
{
  const int lane = threadIdx.x & 31;
  const int lm   = lane & 15;
  const int hi   = lane >> 4;
  const unsigned short* arow0 = A + (size_t)(mRow + lm) * lda + 8 * hi;
  const unsigned short* arow1 = arow0 + (size_t)16 * lda;
  const unsigned short* b0    = BT + (size_t)(nBase + lm) * ldb + 8 * hi;
  const size_t bstep = (size_t)16 * ldb;
  for (int k0 = 0; k0 < K; k0 += 32) {
    if (k0 + 128 < K) {
      __builtin_prefetch(arow0 + k0 + 128, 0, 0);
      __builtin_prefetch(arow1 + k0 + 128, 0, 0);
    }
    FragBF a0, a1;
    a0.q[0] = *(const uint4*)(arow0 + k0);        // e=0..7  : K = k0 + 8*hi ..
    a0.q[1] = *(const uint4*)(arow0 + k0 + 16);   // e=8..15 : K = k0+16+8*hi ..
    a1.q[0] = *(const uint4*)(arow1 + k0);
    a1.q[1] = *(const uint4*)(arow1 + k0 + 16);
#pragma unroll
    for (int t = 0; t < 4; ++t) {
      FragBF b;
      const unsigned short* bp = b0 + bstep * t + k0;  // K = k0 + 8*hi + e (e<8), k0 + 16 + 8*hi + (e-8)
      b.q[0] = *(const uint4*)(bp);
      b.q[1] = *(const uint4*)(bp + 16);
      acc[0][t] = WMMA_BF16(a0, b, acc[0][t]);
      acc[1][t] = WMMA_BF16(a1, b, acc[1][t]);
    }
  }
}

// ------------------------------ K0: conversions ----------------------------
__global__ void k_cvt_f32_bf16(const float* __restrict__ in,
                               unsigned short* __restrict__ out, int n) {
  int i = blockIdx.x * blockDim.x + threadIdx.x;
  if (i < n) out[i] = f2bf(in[i]);
}

// in[rows_in][cols_in] (f32) -> out[cols_in][rows_in] (bf16)
__global__ void k_transpose_f32_bf16(const float* __restrict__ in,
                                     unsigned short* __restrict__ out,
                                     int rows_in, int cols_in) {
  int i = blockIdx.x * blockDim.x + threadIdx.x;
  if (i >= rows_in * cols_in) return;
  int c_out = i % rows_in;          // row of input
  int r_out = i / rows_in;          // col of input
  out[i] = f2bf(in[(size_t)c_out * cols_in + r_out]);
}

// ------------------------- K1: hidden projection ---------------------------
__global__ void k_gemm_hidden(const unsigned short* __restrict__ xb,
                              const unsigned short* __restrict__ whT,
                              const float* __restrict__ b_hidden,
                              unsigned short* __restrict__ vT,
                              unsigned short* __restrict__ gate)
{
  const int wave = threadIdx.x >> 5, lane = threadIdx.x & 31;
  const int lm = lane & 15, hi = lane >> 4;
  const int mRow  = blockIdx.x * 256 + wave * 32;
  const int nBase = blockIdx.y * 64;
  v8f acc[2][4] = {};
  wmma_tile_32x64(xb, 512, whT, 512, 512, mRow, nBase, acc);

  const int batch = mRow >> 12;
#pragma unroll
  for (int a = 0; a < 2; ++a) {
    const int mA   = mRow + 16 * a;
    const int mloc = (mA & 4095) + 8 * hi;
#pragma unroll
    for (int t = 0; t < 4; ++t) {
      const int nH   = nBase + 16 * t + lm;
      const float bs = b_hidden[nH];
      if (nH < 1024) {
        Pack8 p;
#pragma unroll
        for (int r = 0; r < 8; ++r) p.s[r] = f2bf(acc[a][t][r] + bs);
        *(uint4*)(vT + ((size_t)batch * 1024 + nH) * 4096 + mloc) = p.q;
      } else {
        const int h = nH - 1024;
#pragma unroll
        for (int r = 0; r < 8; ++r) {
          int m = mA + r + 8 * hi;
          gate[(size_t)m * 1024 + h] = f2bf(acc[a][t][r] + bs);
        }
      }
    }
  }
}

// ----------------------------- K2: q/k projection --------------------------
__global__ void k_gemm_qk(const unsigned short* __restrict__ xb,
                          const unsigned short* __restrict__ wqkT,
                          const float* __restrict__ qg, const float* __restrict__ qb,
                          const float* __restrict__ kg, const float* __restrict__ kb,
                          unsigned short* __restrict__ qo,
                          unsigned short* __restrict__ ko)
{
  const int wave = threadIdx.x >> 5, lane = threadIdx.x & 31;
  const int lm = lane & 15, hi = lane >> 4;
  const int mRow  = blockIdx.x * 256 + wave * 32;
  const int nBase = blockIdx.y * 64;
  v8f acc[2][4] = {};
  wmma_tile_32x64(xb, 512, wqkT, 512, 512, mRow, nBase, acc);
#pragma unroll
  for (int a = 0; a < 2; ++a) {
#pragma unroll
    for (int t = 0; t < 4; ++t) {
      const int d = nBase + 16 * t + lm;
      const float qgv = qg[d], qbv = qb[d], kgv = kg[d], kbv = kb[d];
#pragma unroll
      for (int r = 0; r < 8; ++r) {
        const int m = mRow + 16 * a + r + 8 * hi;
        const float c = acc[a][t][r];
        qo[(size_t)m * 128 + d] = f2bf(c * qgv + qbv);
        ko[(size_t)m * 128 + d] = f2bf(c * kgv + kbv);
      }
    }
  }
}

// -------------------- K3: sim = relu(q k^T * scale)^2 ----------------------
__global__ void k_gemm_sim(const unsigned short* __restrict__ q,
                           const unsigned short* __restrict__ k,
                           unsigned short* __restrict__ attn)
{
  const int wave = threadIdx.x >> 5, lane = threadIdx.x & 31;
  const int lm = lane & 15, hi = lane >> 4;
  const int batch = blockIdx.z;
  const unsigned short* A  = q + (size_t)batch * 4096 * 128;
  const unsigned short* BT = k + (size_t)batch * 4096 * 128;
  const int mRow  = blockIdx.x * 256 + wave * 32;
  const int nBase = blockIdx.y * 64;
  v8f acc[2][4] = {};
  wmma_tile_32x64(A, 128, BT, 128, 128, mRow, nBase, acc);
  const float scale = 0.08838834764831845f;  // 128^-0.5
  unsigned short* out = attn + (size_t)batch * 4096 * 4096;
#pragma unroll
  for (int a = 0; a < 2; ++a) {
#pragma unroll
    for (int t = 0; t < 4; ++t) {
      const int j = nBase + 16 * t + lm;
#pragma unroll
      for (int r = 0; r < 8; ++r) {
        const int i = mRow + 16 * a + r + 8 * hi;
        float s = acc[a][t][r] * scale;
        s = (s > 0.0f) ? s * s : 0.0f;   // relu^2
        out[(size_t)i * 4096 + j] = f2bf(s);
      }
    }
  }
}

// --------------------- K4: o = (attn @ v) * gate ---------------------------
// B panel (64 cols of vT) is shared by all 8 waves -> stage through LDS with
// double-buffered global_load_async_to_lds_b128 (ASYNCcnt) + ds_load frags.
// LDS layout: [col][72 halfs] (pad 64->72 keeps b128 align, kills bank conflicts)
#define PV_KC   64                 // K elements per staged chunk
#define PV_PITCH 72                // padded halfs per column
__global__ void k_gemm_pv(const unsigned short* __restrict__ attn,
                          const unsigned short* __restrict__ vT,
                          const unsigned short* __restrict__ gate,
                          unsigned short* __restrict__ o)
{
  __shared__ unsigned short sB[2][64 * PV_PITCH];

  const int tid  = threadIdx.x;
  const int wave = tid >> 5, lane = tid & 31;
  const int lm = lane & 15, hi = lane >> 4;
  const int batch = blockIdx.z;
  const unsigned short* A  = attn + (size_t)batch * 4096 * 4096;
  const unsigned short* BT = vT   + (size_t)batch * 1024 * 4096;
  const int mRow = blockIdx.x * 256 + wave * 32;
  const int n0   = blockIdx.y * 64;

  // Producer mapping: 256 threads x 16B cover 32 cols/call; 2 calls per chunk.
  const int pcol = tid >> 3;              // 0..31
  const int poff = (tid & 7) * 8;         // halfs inside the 64-half K chunk
  const unsigned short* gsrc0 = BT + (size_t)(n0 + pcol)      * 4096 + poff;
  const unsigned short* gsrc1 = BT + (size_t)(n0 + pcol + 32) * 4096 + poff;
  const unsigned lbase0 = (unsigned)(size_t)&sB[0][pcol * PV_PITCH + poff];
  const unsigned lbase1 = (unsigned)(size_t)&sB[0][(pcol + 32) * PV_PITCH + poff];
  const unsigned lstep  = (unsigned)(64 * PV_PITCH * 2);  // bytes between buffers

  v8f acc[2][4] = {};
  const unsigned short* arow0 = A + (size_t)(mRow + lm) * 4096 + 8 * hi;
  const unsigned short* arow1 = arow0 + (size_t)16 * 4096;

  // prologue: stage chunk 0 into buffer 0
  {
    unsigned l0 = lbase0, l1 = lbase1;
    asm volatile("global_load_async_to_lds_b128 %0, %1, off"
                 :: "v"(l0), "v"(gsrc0) : "memory");
    asm volatile("global_load_async_to_lds_b128 %0, %1, off"
                 :: "v"(l1), "v"(gsrc1) : "memory");
  }

  const int NCHUNK = 4096 / PV_KC;       // 64
  for (int c = 0; c < NCHUNK; ++c) {
    const int k0 = c * PV_KC;
    if (c + 1 < NCHUNK) {                // stage next chunk into other buffer
      const unsigned sel = (unsigned)((c + 1) & 1) * lstep;
      unsigned l0 = lbase0 + sel, l1 = lbase1 + sel;
      asm volatile("global_load_async_to_lds_b128 %0, %1, off"
                   :: "v"(l0), "v"(gsrc0 + k0 + PV_KC) : "memory");
      asm volatile("global_load_async_to_lds_b128 %0, %1, off"
                   :: "v"(l1), "v"(gsrc1 + k0 + PV_KC) : "memory");
      asm volatile("s_wait_asynccnt 0x2" ::: "memory");  // chunk c landed
    } else {
      asm volatile("s_wait_asynccnt 0x0" ::: "memory");
    }
    __syncthreads();                     // every wave's staging complete

    const unsigned short* sbuf = &sB[c & 1][0];
#pragma unroll
    for (int kk = 0; kk < PV_KC; kk += 32) {
      FragBF a0, a1;
      a0.q[0] = *(const uint4*)(arow0 + k0 + kk);
      a0.q[1] = *(const uint4*)(arow0 + k0 + kk + 16);
      a1.q[0] = *(const uint4*)(arow1 + k0 + kk);
      a1.q[1] = *(const uint4*)(arow1 + k0 + kk + 16);
#pragma unroll
      for (int t = 0; t < 4; ++t) {
        FragBF b;
        const unsigned short* bp = sbuf + (16 * t + lm) * PV_PITCH + kk + 8 * hi;
        b.q[0] = *(const uint4*)(bp);
        b.q[1] = *(const uint4*)(bp + 16);
        acc[0][t] = WMMA_BF16(a0, b, acc[0][t]);
        acc[1][t] = WMMA_BF16(a1, b, acc[1][t]);
      }
    }
    __syncthreads();                     // buffer free before re-staging
  }

#pragma unroll
  for (int a = 0; a < 2; ++a) {
#pragma unroll
    for (int t = 0; t < 4; ++t) {
      const int h = n0 + 16 * t + lm;
#pragma unroll
      for (int r = 0; r < 8; ++r) {
        const int m = mRow + 16 * a + r + 8 * hi;
        const size_t idx = ((size_t)batch * 4096 + m) * 1024 + h;
        o[idx] = f2bf(acc[a][t][r] * bf2f(gate[idx]));
      }
    }
  }
}

// --------------- K5: out = o @ Wo + b_out + x (residual) -------------------
__global__ void k_gemm_out(const unsigned short* __restrict__ o,
                           const unsigned short* __restrict__ woT,
                           const float* __restrict__ b_out,
                           const float* __restrict__ x,
                           float* __restrict__ out)
{
  const int wave = threadIdx.x >> 5, lane = threadIdx.x & 31;
  const int lm = lane & 15, hi = lane >> 4;
  const int mRow  = blockIdx.x * 256 + wave * 32;
  const int nBase = blockIdx.y * 64;
  v8f acc[2][4] = {};
  wmma_tile_32x64(o, 1024, woT, 1024, 1024, mRow, nBase, acc);
#pragma unroll
  for (int a = 0; a < 2; ++a) {
#pragma unroll
    for (int t = 0; t < 4; ++t) {
      const int d = nBase + 16 * t + lm;
      const float bs = b_out[d];
#pragma unroll
      for (int r = 0; r < 8; ++r) {
        const int m = mRow + 16 * a + r + 8 * hi;
        const size_t idx = (size_t)m * 512 + d;
        out[idx] = acc[a][t][r] + bs + x[idx];
      }
    }
  }
}

// ---------------------------------------------------------------------------
extern "C" void kernel_launch(void* const* d_in, const int* in_sizes, int n_in,
                              void* d_out, int out_size, void* d_ws, size_t ws_size,
                              hipStream_t stream) {
  const float* x        = (const float*)d_in[0];
  const float* w_hidden = (const float*)d_in[1];
  const float* b_hidden = (const float*)d_in[2];
  const float* w_qk     = (const float*)d_in[3];
  const float* q_gamma  = (const float*)d_in[4];
  const float* q_beta   = (const float*)d_in[5];
  const float* k_gamma  = (const float*)d_in[6];
  const float* k_beta   = (const float*)d_in[7];
  const float* w_out    = (const float*)d_in[8];
  const float* b_out    = (const float*)d_in[9];
  float* out = (float*)d_out;

  char* ws = (char*)d_ws;
  size_t off = 0;
  auto alloc = [&](size_t bytes) -> void* {
    void* p = ws + off;
    off = (off + bytes + 255) & ~(size_t)255;
    return p;
  };
  unsigned short* xb   = (unsigned short*)alloc(16384ull * 512 * 2);
  unsigned short* whT  = (unsigned short*)alloc(2048ull * 512 * 2);
  unsigned short* wqkT = (unsigned short*)alloc(128ull * 512 * 2);
  unsigned short* woT  = (unsigned short*)alloc(512ull * 1024 * 2);
  unsigned short* qbuf = (unsigned short*)alloc(16384ull * 128 * 2);
  unsigned short* kbuf = (unsigned short*)alloc(16384ull * 128 * 2);
  unsigned short* vT   = (unsigned short*)alloc(4ull * 1024 * 4096 * 2);
  unsigned short* gate = (unsigned short*)alloc(16384ull * 1024 * 2);
  unsigned short* attn = (unsigned short*)alloc(4ull * 4096 * 4096 * 2);
  unsigned short* obuf = (unsigned short*)alloc(16384ull * 1024 * 2);

  // K0: precision conversion + weight transposes (to [N][K] bf16)
  const int nx = 16384 * 512;
  k_cvt_f32_bf16<<<(nx + 255) / 256, 256, 0, stream>>>(x, xb, nx);
  k_transpose_f32_bf16<<<(512 * 2048 + 255) / 256, 256, 0, stream>>>(w_hidden, whT, 512, 2048);
  k_transpose_f32_bf16<<<(512 * 128  + 255) / 256, 256, 0, stream>>>(w_qk,     wqkT, 512, 128);
  k_transpose_f32_bf16<<<(1024 * 512 + 255) / 256, 256, 0, stream>>>(w_out,    woT, 1024, 512);

  // K1..K5: WMMA GEMM chain
  k_gemm_hidden<<<dim3(64, 32), 256, 0, stream>>>(xb, whT, b_hidden, vT, gate);
  k_gemm_qk    <<<dim3(64,  2), 256, 0, stream>>>(xb, wqkT, q_gamma, q_beta,
                                                  k_gamma, k_beta, qbuf, kbuf);
  k_gemm_sim   <<<dim3(16, 64, 4), 256, 0, stream>>>(qbuf, kbuf, attn);
  k_gemm_pv    <<<dim3(16, 16, 4), 256, 0, stream>>>(attn, vT, gate, obuf);
  k_gemm_out   <<<dim3(64,  8), 256, 0, stream>>>(obuf, woT, b_out, x, out);
}
